// Decoder_85899346625
// MI455X (gfx1250) — matrix operand order repaired and checked by reading, not compiled, not run on hardware
//
#include <hip/hip_runtime.h>

// MI455X / gfx1250, wave32. All GEMMs via v_wmma_f32_16x16x32_bf16.
// Tile staging uses CDNA5 async-to-LDS (ASYNCcnt) where the copy is a pure
// 16B-per-lane move; fp32 weights are converted in-VALU and ds_store'd.
// Decoder: 6 layers of (causal self-attn, cross-attn, FFN) with add+LN.

typedef __attribute__((ext_vector_type(16))) __bf16 v16bf;
typedef __attribute__((ext_vector_type(8)))  __bf16 v8bf;
typedef __attribute__((ext_vector_type(8)))  float  v8f;

#define USE_ASYNC 1

static __device__ __forceinline__ __bf16 f2bf(float f) {
  // round-to-nearest-even fp32 -> bf16
  unsigned u = __builtin_bit_cast(unsigned, f);
  unsigned r = u + 0x7fffu + ((u >> 16) & 1u);
  unsigned short h = (unsigned short)(r >> 16);
  return __builtin_bit_cast(__bf16, h);
}

static __device__ __forceinline__ unsigned lds_u32(const void* p) {
  // low 32 bits of the generic (shared-aperture) address == LDS byte offset
  return (unsigned)(unsigned long long)p;
}

#if USE_ASYNC
static __device__ __forceinline__ void async_ld_b128(unsigned lds_off, const void* gptr) {
  asm volatile("global_load_async_to_lds_b128 %0, %1, off"
               :: "v"(lds_off), "v"(gptr) : "memory");
}
static __device__ __forceinline__ void wait_async0() {
  asm volatile("s_wait_asynccnt 0x0" ::: "memory");
}
#endif

#define TM 128
#define TN 128
#define TK 32
#define LDA_S 40   // padded LDS row stride (bf16 elems): 80B rows keep 16B alignment
#define LDB_S 40

// C[M,N] = alpha * A[M,K](bf16) x B + bias
// BMODE 0: B = fp32 [K,N] row-major (weights, converted in-kernel)
// BMODE 1: B = bf16 [K,N] row-major (probs @ V)
// BMODE 2: B = bf16 [N,K] row-major (Q @ K^T scores; row n of B is column n)
// causal!=0: skip output tiles entirely above the diagonal.
template<int BMODE, bool OUT_BF16, bool HAS_BIAS>
__global__ __launch_bounds__(256) void gemm_k(
    const __bf16* __restrict__ A, const void* __restrict__ Bsrc,
    const float* __restrict__ bias, void* __restrict__ Cdst,
    int M, int N, int K,
    long long sA, long long sB, long long sC,
    float alpha, int causal)
{
  __shared__ __bf16 As[TM * LDA_S];
  __shared__ __bf16 Bs[TN * LDB_S];   // transposed: Bs[n][k]

  const int bz = blockIdx.z;
  A += (long long)bz * sA;
  const float*  Bf = (const float*)Bsrc  + ((BMODE == 0) ? (long long)bz * sB : 0);
  const __bf16* Bh = (const __bf16*)Bsrc + ((BMODE != 0) ? (long long)bz * sB : 0);

  const int n0 = blockIdx.x * TN;
  const int m0 = blockIdx.y * TM;
  if (causal && n0 > m0 + TM - 1) return;   // fully-masked score tile

  const int tid  = threadIdx.x;
  const int lane = tid & 31;
  const int wv   = tid >> 5;
  const int wm   = wv & 3;    // 4 wave-rows of 32
  const int wn   = wv >> 2;   // 2 wave-cols of 64
  const int hi   = lane >> 4;
  const int ln   = lane & 15;

  v8f c00 = {}; v8f c01 = {}; v8f c02 = {}; v8f c03 = {};
  v8f c10 = {}; v8f c11 = {}; v8f c12 = {}; v8f c13 = {};

  // A staging map: 128 rows x 32 k, 16 bf16 (32B) per thread, contiguous
  const int arow = tid >> 1;        // 0..127
  const int aoff = (tid & 1) * 16;  // 0 or 16 (bf16 elems)
  const __bf16* gA = A + (long long)(m0 + arow) * K + aoff;
  const unsigned ldsA = lds_u32(&As[arow * LDA_S + aoff]);

  // B staging maps
  const int kr = tid >> 3;          // 0..31   (BMODE 0/1)
  const int nc = (tid & 7) * 16;    //         (BMODE 0/1: 16 cols per thread)
  const int nr = tid >> 1;          // 0..127  (BMODE 2)
  const int kc = (tid & 1) * 16;    //         (BMODE 2)
  const unsigned ldsB2 = lds_u32(&Bs[nr * LDB_S + kc]);

  for (int k0 = 0; k0 < K; k0 += TK) {
    // ---- stage A tile (bf16 row-major, contiguous) ----
#if USE_ASYNC
    async_ld_b128(ldsA,      gA + k0);
    async_ld_b128(ldsA + 16, gA + k0 + 8);
#else
    {
      const float4* s = (const float4*)(gA + k0);
      float4 p0 = s[0], p1 = s[1];
      *(float4*)&As[arow * LDA_S + aoff]     = p0;
      *(float4*)&As[arow * LDA_S + aoff + 8] = p1;
    }
#endif
    // ---- stage B tile transposed into Bs[n][k] ----
    if (BMODE == 0) {
      const float4* s = (const float4*)(Bf + (long long)(k0 + kr) * N + n0 + nc);
      float4 p0 = s[0], p1 = s[1], p2 = s[2], p3 = s[3];
      float vals[16] = {p0.x, p0.y, p0.z, p0.w, p1.x, p1.y, p1.z, p1.w,
                        p2.x, p2.y, p2.z, p2.w, p3.x, p3.y, p3.z, p3.w};
#pragma unroll
      for (int j = 0; j < 16; ++j) Bs[(nc + j) * LDB_S + kr] = f2bf(vals[j]);
      if (k0 + TK < K)
        __builtin_prefetch(Bf + (long long)(k0 + TK + kr) * N + n0 + nc, 0, 3);
    } else if (BMODE == 1) {
      const v8bf* s = (const v8bf*)(Bh + (long long)(k0 + kr) * N + n0 + nc);
      v8bf p0 = s[0], p1 = s[1];
#pragma unroll
      for (int j = 0; j < 8; ++j) {
        Bs[(nc + j)     * LDB_S + kr] = p0[j];
        Bs[(nc + j + 8) * LDB_S + kr] = p1[j];
      }
      if (k0 + TK < K)
        __builtin_prefetch(Bh + (long long)(k0 + TK + kr) * N + n0 + nc, 0, 3);
    } else {
      const __bf16* gB = Bh + (long long)(n0 + nr) * K + k0 + kc;
#if USE_ASYNC
      async_ld_b128(ldsB2,      gB);
      async_ld_b128(ldsB2 + 16, gB + 8);
#else
      float4 p0 = ((const float4*)gB)[0], p1 = ((const float4*)gB)[1];
      *(float4*)&Bs[nr * LDB_S + kc]     = p0;
      *(float4*)&Bs[nr * LDB_S + kc + 8] = p1;
#endif
    }
#if USE_ASYNC
    wait_async0();
#endif
    __syncthreads();

    // ---- per-lane fragment gather (ISA 7.12.2 layouts) ----
    // A 16x32 bf16: lane row = M; elems K = e + 8*hi (e<8), 8 + e + 8*hi (e>=8)
    v16bf a0, a1;
    {
      const __bf16* pr0 = &As[(wm * 32 + 0 * 16 + ln) * LDA_S];
      const __bf16* pr1 = &As[(wm * 32 + 1 * 16 + ln) * LDA_S];
      v8bf l0 = *(const v8bf*)(pr0 + 8 * hi), h0 = *(const v8bf*)(pr0 + 16 + 8 * hi);
      v8bf l1 = *(const v8bf*)(pr1 + 8 * hi), h1 = *(const v8bf*)(pr1 + 16 + 8 * hi);
#pragma unroll
      for (int e = 0; e < 8; ++e) {
        a0[e] = l0[e]; a0[e + 8] = h0[e];
        a1[e] = l1[e]; a1[e + 8] = h1[e];
      }
    }
    // B 32x16 bf16: lane col = N; elems K = e + 8*hi (e<8), 8 + e + 8*hi (e>=8)
    v16bf b0, b1, b2, b3;
#define LOADB(BV, F)                                                          \
    {                                                                         \
      const __bf16* pc = &Bs[(wn * 64 + (F) * 16 + ln) * LDB_S + 8 * hi];     \
      v8bf lo = *(const v8bf*)pc, h8 = *(const v8bf*)(pc + 16);               \
      _Pragma("unroll")                                                       \
      for (int e = 0; e < 8; ++e) { BV[e] = lo[e]; BV[e + 8] = h8[e]; }       \
    }
    LOADB(b0, 0) LOADB(b1, 1) LOADB(b2, 2) LOADB(b3, 3)
#undef LOADB

    c00 = __builtin_amdgcn_wmma_f32_16x16x32_bf16(false, a0, false, b0, (short)0, c00, false, false);
    c01 = __builtin_amdgcn_wmma_f32_16x16x32_bf16(false, a0, false, b1, (short)0, c01, false, false);
    c02 = __builtin_amdgcn_wmma_f32_16x16x32_bf16(false, a0, false, b2, (short)0, c02, false, false);
    c03 = __builtin_amdgcn_wmma_f32_16x16x32_bf16(false, a0, false, b3, (short)0, c03, false, false);
    c10 = __builtin_amdgcn_wmma_f32_16x16x32_bf16(false, a1, false, b0, (short)0, c10, false, false);
    c11 = __builtin_amdgcn_wmma_f32_16x16x32_bf16(false, a1, false, b1, (short)0, c11, false, false);
    c12 = __builtin_amdgcn_wmma_f32_16x16x32_bf16(false, a1, false, b2, (short)0, c12, false, false);
    c13 = __builtin_amdgcn_wmma_f32_16x16x32_bf16(false, a1, false, b3, (short)0, c13, false, false);

    __syncthreads();
  }

  // ---- epilogue: C/D layout -> row = r + 8*hi, col = ln (per 16x16 tile) ----
  const long long coff = (long long)bz * sC;
  float*  Cf = (float*)Cdst;
  __bf16* Cb = (__bf16*)Cdst;

#define EPI(CC, FM, FN)                                                        \
  {                                                                            \
    int col = n0 + wn * 64 + (FN) * 16 + ln;                                   \
    float bv = HAS_BIAS ? bias[col] : 0.f;                                     \
    _Pragma("unroll")                                                          \
    for (int r = 0; r < 8; ++r) {                                              \
      int row = m0 + wm * 32 + (FM) * 16 + r + 8 * hi;                         \
      float v = CC[r] * alpha + bv;                                            \
      long long idx = coff + (long long)row * N + col;                         \
      if (OUT_BF16) Cb[idx] = f2bf(v); else Cf[idx] = v;                       \
    }                                                                          \
  }
  EPI(c00, 0, 0) EPI(c01, 0, 1) EPI(c02, 0, 2) EPI(c03, 0, 3)
  EPI(c10, 1, 0) EPI(c11, 1, 1) EPI(c12, 1, 2) EPI(c13, 1, 3)
#undef EPI
}

// Row softmax over fp32 scores -> bf16 probs. causal: row i keeps j<=i, zeros elsewhere.
__global__ __launch_bounds__(256) void softmax_k(const float* __restrict__ Sc,
                                                 __bf16* __restrict__ P,
                                                 int n, int causal)
{
  __shared__ float red[256];
  const int i = blockIdx.x;
  const long long base = ((long long)blockIdx.y * gridDim.x + i) * n;
  const int valid = causal ? (i + 1) : n;
  const int tid = threadIdx.x;

  float m = -3.402823466e38f;
  for (int j = tid; j < valid; j += 256) m = fmaxf(m, Sc[base + j]);
  red[tid] = m; __syncthreads();
  for (int s = 128; s > 0; s >>= 1) { if (tid < s) red[tid] = fmaxf(red[tid], red[tid + s]); __syncthreads(); }
  const float mx = red[0];
  __syncthreads();

  float sum = 0.f;
  for (int j = tid; j < valid; j += 256) sum += __expf(Sc[base + j] - mx);
  red[tid] = sum; __syncthreads();
  for (int s = 128; s > 0; s >>= 1) { if (tid < s) red[tid] += red[tid + s]; __syncthreads(); }
  const float inv = 1.f / red[0];

  for (int j = tid; j < n; j += 256)
    P[base + j] = (j < valid) ? f2bf(__expf(Sc[base + j] - mx) * inv) : f2bf(0.f);
}

// x = LN(x + y)*g + b ; writes fp32 (xo) and bf16 (xb). D fixed = 1024, block = 256.
__global__ __launch_bounds__(256) void add_ln_k(const float* __restrict__ x,
                                                const float* __restrict__ y,
                                                const float* __restrict__ g,
                                                const float* __restrict__ b,
                                                float* __restrict__ xo,
                                                __bf16* __restrict__ xb)
{
  const int D = 1024;
  __shared__ float r1[256], r2[256];
  const long long base = (long long)blockIdx.x * D;
  const int tid = threadIdx.x;
  float v[4]; float s = 0.f, ss = 0.f;
#pragma unroll
  for (int u = 0; u < 4; ++u) {
    int j = tid + u * 256;
    float a = x[base + j] + y[base + j];
    v[u] = a; s += a; ss += a * a;
  }
  r1[tid] = s; r2[tid] = ss; __syncthreads();
  for (int t = 128; t > 0; t >>= 1) {
    if (tid < t) { r1[tid] += r1[tid + t]; r2[tid] += r2[tid + t]; }
    __syncthreads();
  }
  const float mean = r1[0] * (1.f / D);
  const float var  = r2[0] * (1.f / D) - mean * mean;
  const float rinv = rsqrtf(var + 1e-5f);
#pragma unroll
  for (int u = 0; u < 4; ++u) {
    int j = tid + u * 256;
    float o = (v[u] - mean) * rinv * g[j] + b[j];
    xo[base + j] = o;
    xb[base + j] = f2bf(o);
  }
}

// fp32 -> (optional fp32 copy, bf16)
__global__ void conv_k(const float* __restrict__ in, float* __restrict__ fo,
                       __bf16* __restrict__ bo, long long n)
{
  long long i = (long long)blockIdx.x * blockDim.x + threadIdx.x;
  if (i >= n) return;
  float v = in[i];
  if (fo) fo[i] = v;
  bo[i] = f2bf(v);
}

extern "C" void kernel_launch(void* const* d_in, const int* in_sizes, int n_in,
                              void* d_out, int out_size, void* d_ws, size_t ws_size,
                              hipStream_t stream)
{
  (void)in_sizes; (void)n_in; (void)out_size; (void)ws_size;
  const int Bn = 4, S = 1024, D = 1024, DFF = 4096, L = 6;
  const int BS = Bn * S;                    // 4096
  const long long BSD = (long long)BS * D;  // 4M

  const float* x_in   = (const float*)d_in[0];
  const float* enc_in = (const float*)d_in[1];
  // d_in[2], d_in[3]: masks — structurally causal / all-true, applied by index.
  const float* Wq1 = (const float*)d_in[4],  *Wk1 = (const float*)d_in[5];
  const float* Wv1 = (const float*)d_in[6],  *Wo1 = (const float*)d_in[7];
  const float* g1  = (const float*)d_in[8],  *b1  = (const float*)d_in[9];
  const float* Wq2 = (const float*)d_in[10], *Wk2 = (const float*)d_in[11];
  const float* Wv2 = (const float*)d_in[12], *Wo2 = (const float*)d_in[13];
  const float* g2  = (const float*)d_in[14], *b2  = (const float*)d_in[15];
  const float* Wf1 = (const float*)d_in[16], *bf1 = (const float*)d_in[17];
  const float* Wf2 = (const float*)d_in[18], *bf2 = (const float*)d_in[19];
  const float* g3  = (const float*)d_in[20], *b3  = (const float*)d_in[21];

  // workspace layout (~136 MB)
  char* w = (char*)d_ws;
  float*  xf   = (float*)w;  w += BSD * 4;                    // fp32 activations
  __bf16* xb   = (__bf16*)w; w += BSD * 2;                    // bf16 activations
  __bf16* encb = (__bf16*)w; w += BSD * 2;                    // bf16 encoder output
  __bf16* qb   = (__bf16*)w; w += BSD * 2;
  __bf16* kb   = (__bf16*)w; w += BSD * 2;
  __bf16* vb   = (__bf16*)w; w += BSD * 2;
  float*  sc   = (float*)w;  w += (long long)Bn * S * S * 4;  // fp32 scores
  __bf16* pb   = (__bf16*)w; w += (long long)Bn * S * S * 2;  // bf16 probs
  __bf16* ab   = (__bf16*)w; w += BSD * 2;                    // bf16 attn out
  float*  yf   = (float*)w;  w += BSD * 4;                    // fp32 branch out
  __bf16* hb   = (__bf16*)w; w += (long long)BS * DFF * 2;    // bf16 FFN hidden

  dim3 blk(256);
  dim3 gProj(D / TN, BS / TM, 1);
  dim3 gScore(S / TN, S / TM, Bn);
  dim3 gPV(D / TN, S / TM, Bn);
  dim3 gF1(DFF / TN, BS / TM, 1);
  dim3 gF2(D / TN, BS / TM, 1);
  const long long sSD = (long long)S * D, sSS = (long long)S * S;

  conv_k<<<(unsigned)((BSD + 255) / 256), blk, 0, stream>>>(x_in, xf, xb, BSD);
  conv_k<<<(unsigned)((BSD + 255) / 256), blk, 0, stream>>>(enc_in, nullptr, encb, BSD);

  for (int l = 0; l < L; ++l) {
    const long long wo_ = (long long)l * D * D;
    // ---------- masked self-attention ----------
    gemm_k<0,true ,false><<<gProj,  blk, 0, stream>>>(xb, Wq1 + wo_, nullptr, qb, BS, D, D, 0, 0, 0, 1.f, 0);
    gemm_k<0,true ,false><<<gProj,  blk, 0, stream>>>(xb, Wk1 + wo_, nullptr, kb, BS, D, D, 0, 0, 0, 1.f, 0);
    gemm_k<0,true ,false><<<gProj,  blk, 0, stream>>>(xb, Wv1 + wo_, nullptr, vb, BS, D, D, 0, 0, 0, 1.f, 0);
    gemm_k<2,false,false><<<gScore, blk, 0, stream>>>(qb, kb, nullptr, sc, S, S, D, sSD, sSD, sSS, 0.125f, 1);
    softmax_k<<<dim3(S, Bn), blk, 0, stream>>>(sc, pb, S, 1);
    gemm_k<1,true ,false><<<gPV,    blk, 0, stream>>>(pb, vb, nullptr, ab, S, D, S, sSS, sSD, sSD, 1.f, 0);
    gemm_k<0,false,false><<<gProj,  blk, 0, stream>>>(ab, Wo1 + wo_, nullptr, yf, BS, D, D, 0, 0, 0, 1.f, 0);
    add_ln_k<<<BS, blk, 0, stream>>>(xf, yf, g1 + l * D, b1 + l * D, xf, xb);
    // ---------- cross-attention ----------
    gemm_k<0,true ,false><<<gProj,  blk, 0, stream>>>(xb,   Wq2 + wo_, nullptr, qb, BS, D, D, 0, 0, 0, 1.f, 0);
    gemm_k<0,true ,false><<<gProj,  blk, 0, stream>>>(encb, Wk2 + wo_, nullptr, kb, BS, D, D, 0, 0, 0, 1.f, 0);
    gemm_k<0,true ,false><<<gProj,  blk, 0, stream>>>(encb, Wv2 + wo_, nullptr, vb, BS, D, D, 0, 0, 0, 1.f, 0);
    gemm_k<2,false,false><<<gScore, blk, 0, stream>>>(qb, kb, nullptr, sc, S, S, D, sSD, sSD, sSS, 0.125f, 0);
    softmax_k<<<dim3(S, Bn), blk, 0, stream>>>(sc, pb, S, 0);
    gemm_k<1,true ,false><<<gPV,    blk, 0, stream>>>(pb, vb, nullptr, ab, S, D, S, sSS, sSD, sSD, 1.f, 0);
    gemm_k<0,false,false><<<gProj,  blk, 0, stream>>>(ab, Wo2 + wo_, nullptr, yf, BS, D, D, 0, 0, 0, 1.f, 0);
    add_ln_k<<<BS, blk, 0, stream>>>(xf, yf, g2 + l * D, b2 + l * D, xf, xb);
    // ---------- feed-forward (no activation) ----------
    gemm_k<0,true ,true ><<<gF1, blk, 0, stream>>>(xb, Wf1 + (long long)l * D * DFF, bf1 + (long long)l * DFF,
                                                   hb, BS, DFF, D, 0, 0, 0, 1.f, 0);
    gemm_k<0,false,true ><<<gF2, blk, 0, stream>>>(hb, Wf2 + (long long)l * DFF * D, bf2 + (long long)l * D,
                                                   yf, BS, D, DFF, 0, 0, 0, 1.f, 0);
    float* xo = (l == L - 1) ? (float*)d_out : xf;
    add_ln_k<<<BS, blk, 0, stream>>>(xf, yf, g3 + l * D, b3 + l * D, xo, xb);
  }
}
